// Model_28174985462163
// MI455X (gfx1250) — compile-verified, matrix operand order checked by reading, not run on hardware
//
#include <hip/hip_runtime.h>
#include <hip/hip_bf16.h>

// ---------------------------------------------------------------------------
// 4-layer LSTM (B=64, T=16, H=2048) for gfx1250.
// GEMMs: v_wmma_f32_16x16x32_f16 (f16 in, f32 accumulate).
// B-operand staged per-block into LDS (async global->LDS, double buffered,
// ASYNCcnt-tracked) to remove 4x cross-wave vmem redundancy.
// ---------------------------------------------------------------------------

typedef __attribute__((ext_vector_type(16))) _Float16 v16h;
typedef __attribute__((ext_vector_type(8)))  _Float16 v8h;
typedef __attribute__((ext_vector_type(8)))  float    v8f;
typedef __attribute__((ext_vector_type(4)))  int      v4i;

#define AS1 __attribute__((address_space(1)))
#define AS3 __attribute__((address_space(3)))

#define HID   2048
#define G4    8192      // 4*HID
#define BSZ   64
#define TSTEP 16
#define RROWS 1024      // TSTEP*BSZ

#if defined(__gfx1250__) && \
    __has_builtin(__builtin_amdgcn_global_load_async_to_lds_b128) && \
    __has_builtin(__builtin_amdgcn_s_wait_asynccnt)
#define USE_ASYNC_LDS 1
#else
#define USE_ASYNC_LDS 0
#endif

union AConcat { v16h v; v8h g[2]; };

__device__ __forceinline__ float sigf(float x)   { return 1.0f / (1.0f + __expf(-x)); }
__device__ __forceinline__ float tanh_f(float x) { return 2.0f * sigf(2.0f * x) - 1.0f; }

// Load the 16-bit A-operand (16x32, M x K) for one wave from a row-major f16
// matrix. Per ISA layout: lane L holds row M=L%16; half e -> K offset
//   k = e + 8*((e>>3) + (L>>4))  => two contiguous 8-half (16B) groups at
//   base = 8*(L>>4) and base+16.
__device__ __forceinline__ v16h load_a_tile(const _Float16* __restrict__ rowPtr) {
    AConcat a;
    a.g[0] = *(const v8h*)(rowPtr);
    a.g[1] = *(const v8h*)(rowPtr + 16);
    return a.v;
}

// Stage one lane-slot (32B) of a B tile into LDS.
__device__ __forceinline__ void stage_b(const _Float16* __restrict__ gsrc,
                                        _Float16* lds) {
#if USE_ASYNC_LDS
    __builtin_amdgcn_global_load_async_to_lds_b128(
        (AS1 v4i*)gsrc, (AS3 v4i*)lds, 0, 0);
    __builtin_amdgcn_global_load_async_to_lds_b128(
        (AS1 v4i*)(gsrc + 8), (AS3 v4i*)(lds + 8), 0, 0);
#else
    *(v16h*)lds = *(const v16h*)gsrc;
#endif
}

// Drain own wave's async copies, then make all waves' copies visible.
__device__ __forceinline__ void stage_wait() {
#if USE_ASYNC_LDS
    __builtin_amdgcn_s_wait_asynccnt(0);
#endif
    __syncthreads();
}

// ---------------------------------------------------------------------------
// Pack a [N4H x K] fp32 weight matrix (row-major, PyTorch w_ih / w_hh) into
// the WMMA B-operand tiling. B = W^T (K x N). Tile (nt, kc): 32 lanes x 16
// halves contiguous; lane L: N = nt*16 + (L&15), half e:
//   K = kc*32 + 8*(L>>4) + (e&7) + 16*(e>>3)
// ---------------------------------------------------------------------------
__global__ void pack_wB(const float* __restrict__ w, _Float16* __restrict__ dst,
                        int K, int totalThreads) {
    int tid = blockIdx.x * blockDim.x + threadIdx.x;
    if (tid >= totalThreads) return;
    int lane = tid & 31;
    int tl   = tid >> 5;
    int kcN  = K >> 5;
    int kc   = tl % kcN;
    int nt   = tl / kcN;
    int n    = (nt << 4) + (lane & 15);
    int kb   = (kc << 5) + ((lane >> 4) << 3);
    const float* src = w + (size_t)n * K + kb;
    v16h out;
#pragma unroll
    for (int e = 0; e < 16; ++e) {
        int koff = (e & 7) + ((e >> 3) << 4);
        out[e] = (_Float16)src[koff];
    }
    *(v16h*)(dst + (size_t)tid * 16) = out;
}

// x [B, T, 1024] fp32 -> f16 row-major [t*B + b][1024]
__global__ void pack_x0(const float* __restrict__ x, _Float16* __restrict__ dst) {
    int tid = blockIdx.x * blockDim.x + threadIdx.x;
    if (tid >= RROWS * 1024) return;
    int d = tid & 1023;
    int r = tid >> 10;
    int t = r >> 6;
    int b = r & 63;
    dst[tid] = (_Float16)x[((size_t)b * TSTEP + t) * 1024 + d];
}

__global__ void zero_state(_Float16* __restrict__ hzero, float* __restrict__ c) {
    int tid = blockIdx.x * blockDim.x + threadIdx.x;
    if (tid < BSZ * HID) { hzero[tid] = (_Float16)0.0f; c[tid] = 0.0f; }
}

__global__ void zero_c(float* __restrict__ c) {
    int tid = blockIdx.x * blockDim.x + threadIdx.x;
    if (tid < BSZ * HID) c[tid] = 0.0f;
}

// ---------------------------------------------------------------------------
// xi[r][0..8191] = X[r][:] @ W_ih^T + b_ih + b_hh     (r = t*B + b, R = 1024)
// grid = (G4/64, R/64), block = 128 (4 waves). Wave w: rows m0 = by*64+w*16,
// four 16-wide column tiles n = bx*64 + j*16 (shared by all 4 waves -> LDS).
// ---------------------------------------------------------------------------
__global__ __launch_bounds__(128)
void xproj_kernel(const _Float16* __restrict__ X, const _Float16* __restrict__ Wpk,
                  const float* __restrict__ b_ih, const float* __restrict__ b_hh,
                  float* __restrict__ xi, int K) {
    __shared__ __align__(32) _Float16 smem[2][4][32][16];   // 8 KB, double buffered

    const int lane = threadIdx.x & 31;
    const int wv   = threadIdx.x >> 5;
    const int m0   = blockIdx.y * 64 + wv * 16;
    const int kcN  = K >> 5;

    v8f acc[4] = {v8f{}, v8f{}, v8f{}, v8f{}};

    const int row = m0 + (lane & 15);
    const _Float16* aRow = X + (size_t)row * K + ((lane >> 4) << 3);

    // this thread's copy slot: tile cg, lane-slot cl (128 threads = 4x32 slots)
    const int cg = threadIdx.x >> 5;
    const int cl = threadIdx.x & 31;
    const _Float16* csrc =
        Wpk + ((size_t)(blockIdx.x * 4 + cg) * kcN * 32 + cl) * 16;

    stage_b(csrc, &smem[0][cg][cl][0]);                     // prologue: kc = 0

    for (int kc = 0; kc < kcN; ++kc) {
        stage_wait();                                       // buf kc ready everywhere
        if (kc + 1 < kcN)
            stage_b(csrc + (size_t)(kc + 1) * 512, &smem[(kc + 1) & 1][cg][cl][0]);
        v16h a = load_a_tile(aRow + kc * 32);
#pragma unroll
        for (int j = 0; j < 4; ++j) {
            v16h b = *(const v16h*)&smem[kc & 1][j][lane][0];
            acc[j] = __builtin_amdgcn_wmma_f32_16x16x32_f16(
                false, a, false, b, (short)0, acc[j], false, false);
        }
    }

    const int nbase = blockIdx.x * 64;
#pragma unroll
    for (int j = 0; j < 4; ++j) {
        int col = nbase + j * 16 + (lane & 15);
        float bias = b_ih[col] + b_hh[col];
#pragma unroll
        for (int r = 0; r < 8; ++r) {
            int mrow = m0 + r + ((lane >> 4) << 3);
            xi[(size_t)mrow * G4 + col] = acc[j][r] + bias;
        }
    }
}

// ---------------------------------------------------------------------------
// Fused recurrent step: gates = h_{t-1} @ W_hh^T + xi[t]; LSTM cell; write h_t.
// grid = H/16 = 128 blocks, block = 128 (4 waves). Block owns hidden slice
// n0 = bx*16 and computes its four gate column tiles (g*H + n0) for all 64
// batch rows (wave w: rows 16w..16w+15). Gate tiles shared by the 4 waves ->
// staged via async LDS. The f32 C/D layout gives each (vgpr r, lane) one
// (b, n) pair consistent across all four accumulators -> cell is pure VALU.
// ---------------------------------------------------------------------------
__global__ __launch_bounds__(128)
void lstm_step(const _Float16* __restrict__ Ain,      // h_{t-1}, [64][H] f16
               const _Float16* __restrict__ Whhpk,    // packed W_hh
               const float* __restrict__ xi, int t,
               float* __restrict__ c,                 // [64][H] f32 (in/out)
               _Float16* __restrict__ seqOut,         // [T*B][H] f16, write row t
               float* __restrict__ finalOut) {        // [B][T][H] f32 or null
    __shared__ __align__(32) _Float16 smem[2][4][32][16];   // 8 KB

    const int lane = threadIdx.x & 31;
    const int wv   = threadIdx.x >> 5;
    const int n0   = blockIdx.x * 16;
    const int kcN  = HID / 32;   // 64

    v8f acc[4] = {v8f{}, v8f{}, v8f{}, v8f{}};

    const int brow = wv * 16 + (lane & 15);
    const _Float16* aRow = Ain + (size_t)brow * HID + ((lane >> 4) << 3);

    const int cg = threadIdx.x >> 5;     // tile (= gate) this thread copies
    const int cl = threadIdx.x & 31;
    const _Float16* csrc =
        Whhpk + ((size_t)(cg * (HID / 16) + blockIdx.x) * kcN * 32 + cl) * 16;

    stage_b(csrc, &smem[0][cg][cl][0]);

    for (int kc = 0; kc < kcN; ++kc) {
        stage_wait();
        if (kc + 1 < kcN)
            stage_b(csrc + (size_t)(kc + 1) * 512, &smem[(kc + 1) & 1][cg][cl][0]);
        v16h a = load_a_tile(aRow + kc * 32);
#pragma unroll
        for (int g = 0; g < 4; ++g) {
            v16h b = *(const v16h*)&smem[kc & 1][g][lane][0];
            acc[g] = __builtin_amdgcn_wmma_f32_16x16x32_f16(
                false, a, false, b, (short)0, acc[g], false, false);
        }
    }

    const int n = n0 + (lane & 15);
#pragma unroll
    for (int r = 0; r < 8; ++r) {
        int b = wv * 16 + r + ((lane >> 4) << 3);
        size_t xr = ((size_t)(t * BSZ + b)) * G4 + n;
        float gi = acc[0][r] + xi[xr];
        float gf = acc[1][r] + xi[xr + HID];
        float gg = acc[2][r] + xi[xr + 2 * HID];
        float go = acc[3][r] + xi[xr + 3 * HID];
        float vi = sigf(gi);
        float vf = sigf(gf);
        float vg = tanh_f(gg);
        float vo = sigf(go);
        size_t cidx = (size_t)b * HID + n;
        float cn = vf * c[cidx] + vi * vg;
        c[cidx] = cn;
        float hn = vo * tanh_f(cn);
        seqOut[((size_t)(t * BSZ + b)) * HID + n] = (_Float16)hn;
        if (finalOut)
            finalOut[((size_t)b * TSTEP + t) * HID + n] = hn;
    }
}

// ---------------------------------------------------------------------------
// Host side
// ---------------------------------------------------------------------------
extern "C" void kernel_launch(void* const* d_in, const int* in_sizes, int n_in,
                              void* d_out, int out_size, void* d_ws, size_t ws_size,
                              hipStream_t stream) {
    const float* x      = (const float*)d_in[0];
    const float* w_ih0  = (const float*)d_in[1];
    const float* w_hh0  = (const float*)d_in[2];
    const float* b_ih0  = (const float*)d_in[3];
    const float* b_hh0  = (const float*)d_in[4];
    const float* w_ih_r = (const float*)d_in[5];
    const float* w_hh_r = (const float*)d_in[6];
    const float* b_ih_r = (const float*)d_in[7];
    const float* b_hh_r = (const float*)d_in[8];

    char* ws = (char*)d_ws;
    auto alloc = [&](size_t bytes) -> char* {
        char* p = ws;
        ws += (bytes + 255) & ~(size_t)255;
        return p;
    };

    _Float16* wih_pk[4];
    _Float16* whh_pk[4];
    wih_pk[0] = (_Float16*)alloc((size_t)G4 * 1024 * 2);
    for (int l = 1; l < 4; ++l) wih_pk[l] = (_Float16*)alloc((size_t)G4 * HID * 2);
    for (int l = 0; l < 4; ++l) whh_pk[l] = (_Float16*)alloc((size_t)G4 * HID * 2);
    float*     xi    = (float*)alloc((size_t)RROWS * G4 * 4);
    _Float16*  seq0  = (_Float16*)alloc((size_t)RROWS * HID * 2);
    _Float16*  seq1  = (_Float16*)alloc((size_t)RROWS * HID * 2);
    _Float16*  hzero = (_Float16*)alloc((size_t)BSZ * HID * 2);
    float*     cbuf  = (float*)alloc((size_t)BSZ * HID * 4);

    // --- one-time packing / init ---
    {
        int n = RROWS * 1024;
        pack_x0<<<(n + 255) / 256, 256, 0, stream>>>(x, seq0);
    }
    {
        int n1 = G4 * 1024 / 16;   // threads for K=1024 pack
        int n2 = G4 * HID / 16;    // threads for K=2048 pack
        pack_wB<<<(n1 + 255) / 256, 256, 0, stream>>>(w_ih0, wih_pk[0], 1024, n1);
        pack_wB<<<(n2 + 255) / 256, 256, 0, stream>>>(w_hh0, whh_pk[0], HID, n2);
        for (int l = 1; l < 4; ++l) {
            const float* wih = w_ih_r + (size_t)(l - 1) * G4 * HID;
            const float* whh = w_hh_r + (size_t)(l - 1) * G4 * HID;
            pack_wB<<<(n2 + 255) / 256, 256, 0, stream>>>(wih, wih_pk[l], HID, n2);
            pack_wB<<<(n2 + 255) / 256, 256, 0, stream>>>(whh, whh_pk[l], HID, n2);
        }
    }
    zero_state<<<(BSZ * HID + 255) / 256, 256, 0, stream>>>(hzero, cbuf);

    // --- layers ---
    const _Float16* seqin = seq0;
    _Float16*       seqout = seq1;
    for (int l = 0; l < 4; ++l) {
        int K = (l == 0) ? 1024 : HID;
        if (l > 0)
            zero_c<<<(BSZ * HID + 255) / 256, 256, 0, stream>>>(cbuf);

        const float* bih = (l == 0) ? b_ih0 : b_ih_r + (size_t)(l - 1) * G4;
        const float* bhh = (l == 0) ? b_hh0 : b_hh_r + (size_t)(l - 1) * G4;

        xproj_kernel<<<dim3(G4 / 64, RROWS / 64), 128, 0, stream>>>(
            seqin, wih_pk[l], bih, bhh, xi, K);

        float* fout = (l == 3) ? (float*)d_out : nullptr;
        for (int t = 0; t < TSTEP; ++t) {
            const _Float16* Ain =
                (t == 0) ? hzero : seqout + (size_t)(t - 1) * BSZ * HID;
            lstm_step<<<HID / 16, 128, 0, stream>>>(
                Ain, whh_pk[l], xi, t, cbuf, seqout, fout);
        }

        // swap sequence buffers for next layer
        _Float16* tmp = (_Float16*)seqin;
        seqin = seqout;
        seqout = tmp;
    }
    (void)in_sizes; (void)n_in; (void)out_size; (void)ws_size;
}
